// QuantizedRNNBase_51513837748434
// MI455X (gfx1250) — matrix operand order repaired and checked by reading, not compiled, not run on hardware
//
#include <hip/hip_runtime.h>
#include <hip/hip_bf16.h>

typedef __attribute__((ext_vector_type(16))) _Float16 v16h;
typedef __attribute__((ext_vector_type(8)))  _Float16 v8h;
typedef __attribute__((ext_vector_type(8)))  float    v8f;

#define T_STEPS 2048
#define BATCH   64
#define IN_DIM  512
#define HID     512
#define G4H     2048
#define NBLK    16   // persistent scan blocks; each owns 32 h-columns

// ---------------------------------------------------------------------------
// CDNA5 async global->LDS copy (ASYNCcnt-tracked path).
// lds_off: raw LDS byte offset (low 32 bits of a generic pointer to LDS,
// per ISA 10.2 aperture rule). 16 bytes per lane per issue.
// Async loads complete in order -> counted waits release partial batches.
// ---------------------------------------------------------------------------
__device__ __forceinline__ void async_copy_g2l(unsigned lds_off, const void* gaddr) {
    asm volatile("global_load_async_to_lds_b128 %0, %1, off"
                 :: "v"(lds_off), "v"(gaddr) : "memory");
}
__device__ __forceinline__ void wait_async_0() {
    asm volatile("s_wait_asynccnt 0x0" ::: "memory");
}
__device__ __forceinline__ void wait_async_8() {   // allow 8 outstanding
    asm volatile("s_wait_asynccnt 0x8" ::: "memory");
}

// ---------------------------------------------------------------------------
// Weight fake-quant: SIGNED_FIXED, scale = 1/sqrt(I+H) = 1/32, 8 bits.
// step = 1/4096 ; q = clamp(round(w*4096), -128, 127)/4096  -> EXACT in fp16.
// ---------------------------------------------------------------------------
__device__ __forceinline__ _Float16 quant_w(float w) {
    float q = rintf(w * 4096.0f);
    q = fminf(fmaxf(q, -128.0f), 127.0f);
    return (_Float16)(q * (1.0f / 4096.0f));
}

// Pack quantized weights W[4H x K] (row-major) into v_wmma_f32_16x16x32_f16
// B-operand fragment order:
//   frag[((ntile*KT + ktile)*32 + lane)*16 + e] =
//       qW[ntile*16 + lane%16, ktile*32 + (lane/16)*8 + (e%8) + (e/8)*16]
__global__ void pack_w_kernel(const float* __restrict__ w,
                              _Float16* __restrict__ frag, int K) {
    int idx = blockIdx.x * blockDim.x + threadIdx.x;     // over 4H*K halves
    if (idx >= G4H * K) return;
    int e    = idx & 15;
    int lane = (idx >> 4) & 31;
    int kt   = (idx >> 9) % (K >> 5);
    int nt   = idx / (K << 4);
    int n = nt * 16 + (lane & 15);
    int k = kt * 32 + ((lane >> 4) << 3) + (e & 7) + ((e >> 3) << 4);
    frag[idx] = quant_w(w[n * K + k]);
}

__global__ void init_kernel(const float* __restrict__ h0,
                            _Float16* __restrict__ hbuf,
                            unsigned* __restrict__ bar) {
    int idx = blockIdx.x * blockDim.x + threadIdx.x;
    if (idx < BATCH * HID) {
        hbuf[idx] = (_Float16)h0[idx];              // ping buffer = h0
        hbuf[BATCH * HID + idx] = (_Float16)0.0f;   // pong buffer
    }
    if (idx == 0) { bar[0] = 0u; bar[1] = 0u; }
}

// ---------------------------------------------------------------------------
// Phase 1: gates_x[T*B, 4H] = x[T*B, I] @ qW_ih^T   (f16 WMMA, f32 accum)
// Block = 8 waves: 128 rows x 64 cols per block. Weights L2-resident;
// throughput phase, latency hidden by occupancy (32 K blocks).
// ---------------------------------------------------------------------------
__global__ void __launch_bounds__(256)
gemm_x_kernel(const float* __restrict__ x,
              const _Float16* __restrict__ wfrag,
              float* __restrict__ gx) {
    const int lane = threadIdx.x & 31;
    const int wave = threadIdx.x >> 5;
    const int row0 = blockIdx.y * 128 + wave * 16;
    const int col0 = blockIdx.x * 64;
    const int abase = ((lane >> 4) & 1) * 8;   // A-frag K sublane offset

    v8f acc[4] = {};
#pragma unroll 4
    for (int kt = 0; kt < IN_DIM / 32; ++kt) {
        // A fragment: lane m holds row (row0 + lane%16),
        // halves 0..7 -> K = kt*32 + abase .. +7 ; halves 8..15 -> +16..+23
        const float* xp = x + (size_t)(row0 + (lane & 15)) * IN_DIM + kt * 32 + abase;
        v16h af;
#pragma unroll
        for (int j = 0; j < 8; ++j) {
            af[j]     = (_Float16)xp[j];       // f32 -> f16 on the fly
            af[8 + j] = (_Float16)xp[16 + j];
        }
        v16h bf[4];
#pragma unroll
        for (int nt = 0; nt < 4; ++nt)
            bf[nt] = *(const v16h*)(wfrag +
                ((size_t)(((col0 >> 4) + nt) * (IN_DIM / 32) + kt) << 9) + lane * 16);
#pragma unroll
        for (int nt = 0; nt < 4; ++nt)
            acc[nt] = __builtin_amdgcn_wmma_f32_16x16x32_f16(
                false, af, false, bf[nt], (short)0, acc[nt], false, false);
    }
    // C layout: VGPR r -> (M = r + 8*(lane>=16), N = lane%16)
#pragma unroll
    for (int nt = 0; nt < 4; ++nt) {
        int row = row0 + ((lane >> 4) & 1) * 8;
        int col = col0 + nt * 16 + (lane & 15);
#pragma unroll
        for (int r = 0; r < 8; ++r)
            gx[(size_t)(row + r) * G4H + col] = acc[nt][r];
    }
}

// ---------------------------------------------------------------------------
// Grid-wide sense/generation barrier over NBLK persistent blocks.
// ---------------------------------------------------------------------------
__device__ __forceinline__ void grid_barrier(unsigned* bar) {
    __syncthreads();
    if (threadIdx.x == 0) {
        volatile unsigned* vgen = bar + 1;
        unsigned g = *vgen;
        __threadfence();
        if (atomicAdd(bar, 1u) == NBLK - 1) {
            atomicExch(bar, 0u);
            __threadfence();
            atomicAdd(bar + 1, 1u);
        } else {
            while (*vgen == g) __builtin_amdgcn_s_sleep(1);
        }
    }
    __syncthreads();
}

// ---------------------------------------------------------------------------
// Phase 2: persistent recurrent scan. Block bid owns h-columns
// [bid*32, bid*32+32) and the matching 4 gate-column groups (i/f/g/o).
// Per step: async-stage h(t) (64 KB) and this step's gates_x slice (32 KB)
// into LDS; counted wait (asynccnt<=8) releases the GEMM as soon as h is
// resident while the gates_x copy overlaps the WMMA work. W_hh slice
// (128 KB) staged once. Static LDS = 256 KB (< 320 KB/WGP -> 1 block/WGP).
// ---------------------------------------------------------------------------
__global__ void __launch_bounds__(256)
lstm_scan_kernel(const float* __restrict__ gx,
                 const _Float16* __restrict__ whfrag,
                 const float* __restrict__ b_ih,
                 const float* __restrict__ b_hh,
                 const float* __restrict__ c0,
                 _Float16* __restrict__ hbuf,
                 unsigned* __restrict__ bar,
                 float* __restrict__ out) {
    __shared__ _Float16 sh_w[8 * 8192];    // 128 KB: block's W_hh fragments
    __shared__ _Float16 sh_h[BATCH * HID]; //  64 KB: staged h(t)
    __shared__ float    gl[64 * 128];      //  32 KB: gates (recurrent part)
    __shared__ float    sh_gx[64 * 128];   //  32 KB: staged gates_x slice

    const int tid  = threadIdx.x;
    const int lane = tid & 31;
    const int wave = tid >> 5;
    const int bid  = blockIdx.x;
    // GEMM ownership: wave -> (mtile, 4 ntiles)
    const int mtile  = wave >> 1;
    const int ntbase = (wave & 1) * 4;
    const int abase  = ((lane >> 4) & 1) * 8;
    // elementwise ownership: column jj, 8 batch rows
    const int jj    = tid & 31;
    const int rb    = (tid >> 5) * 8;
    const int col_h = bid * 32 + jj;

    const unsigned lds_w  = (unsigned)(uintptr_t)(void*)sh_w;
    const unsigned lds_h  = (unsigned)(uintptr_t)(void*)sh_h;
    const unsigned lds_gx = (unsigned)(uintptr_t)(void*)sh_gx;

    // Stage this block's W_hh fragment slice into LDS (once, async).
    // Global ntile ntg's fragments are contiguous: 8192 halves @ ntg*8192.
#pragma unroll
    for (int ntl = 0; ntl < 8; ++ntl) {
        int q   = ntl >> 1;
        int ntg = q * 32 + bid * 2 + (ntl & 1);
        const _Float16* src = whfrag + (size_t)ntg * 8192;
#pragma unroll
        for (int i = 0; i < 4; ++i) {
            int c = i * 256 + tid;                 // 16-byte chunk 0..1023
            async_copy_g2l(lds_w + (unsigned)(ntl * 16384 + c * 16), src + c * 8);
        }
    }

    float bias[4];
#pragma unroll
    for (int q = 0; q < 4; ++q)
        bias[q] = b_ih[q * HID + col_h] + b_hh[q * HID + col_h];

    float creg[8], hreg[8];
#pragma unroll
    for (int k = 0; k < 8; ++k) {
        creg[k] = c0[(rb + k) * HID + col_h];
        hreg[k] = 0.0f;
    }

#pragma unroll 1
    for (int t = 0; t < T_STEPS; ++t) {
        // (1) stage h(t): 4096 x 16B chunks, 16 per thread
        const _Float16* hsrc = hbuf + (t & 1) * (BATCH * HID);
#pragma unroll
        for (int i = 0; i < 16; ++i) {
            int c = i * 256 + tid;
            async_copy_g2l(lds_h + (unsigned)(c * 16), hsrc + c * 8);
        }
        // (2) stage gates_x slice [64 rows x 4 gates x 32 cols]: 8 chunks/thread
        //     chunk c = b*32 + q*8 + s  -> 16B @ gxt[b*4H + q*512 + s*4]
        const float* gxt = gx + (size_t)t * (BATCH * G4H) + bid * 32;
#pragma unroll
        for (int i = 0; i < 8; ++i) {
            int c = i * 256 + tid;
            int b = c >> 5, q = (c >> 3) & 3, s = c & 7;
            async_copy_g2l(lds_gx + (unsigned)(c * 16),
                           gxt + (size_t)b * G4H + q * 512 + s * 4);
        }
        // release GEMM once h (first 16/thread, plus W at t==0) has landed;
        // the 8 gates_x chunks/thread keep streaming underneath the WMMAs
        wait_async_8();
        __syncthreads();

        v8f acc[4] = {};
#pragma unroll 2
        for (int kt = 0; kt < HID / 32; ++kt) {
            const _Float16* hp = sh_h + (mtile * 16 + (lane & 15)) * HID
                               + kt * 32 + abase;
            const v8h a0 = *(const v8h*)hp;
            const v8h a1 = *(const v8h*)(hp + 16);
            v16h af;
#pragma unroll
            for (int j = 0; j < 8; ++j) { af[j] = a0[j]; af[8 + j] = a1[j]; }
            v16h bf[4];
#pragma unroll
            for (int nt = 0; nt < 4; ++nt)
                bf[nt] = *(const v16h*)(sh_w + (((ntbase + nt) * 16 + kt) << 9)
                                        + lane * 16);
#pragma unroll
            for (int nt = 0; nt < 4; ++nt)
                acc[nt] = __builtin_amdgcn_wmma_f32_16x16x32_f16(
                    false, af, false, bf[nt], (short)0, acc[nt], false, false);
        }
        // WMMA layout -> natural layout via LDS
#pragma unroll
        for (int nt = 0; nt < 4; ++nt) {
            int ntl = ntbase + nt;
            int row = mtile * 16 + ((lane >> 4) & 1) * 8;
            int col = ntl * 16 + (lane & 15);
#pragma unroll
            for (int r = 0; r < 8; ++r)
                gl[(row + r) * 128 + col] = acc[nt][r];
        }
        wait_async_0();        // gates_x slice fully resident
        __syncthreads();

#pragma unroll
        for (int k = 0; k < 8; ++k) {
            int b = rb + k;
            float vi = gl[b * 128 +  0 + jj] + sh_gx[(b * 4 + 0) * 32 + jj] + bias[0];
            float vf = gl[b * 128 + 32 + jj] + sh_gx[(b * 4 + 1) * 32 + jj] + bias[1];
            float vg = gl[b * 128 + 64 + jj] + sh_gx[(b * 4 + 2) * 32 + jj] + bias[2];
            float vo = gl[b * 128 + 96 + jj] + sh_gx[(b * 4 + 3) * 32 + jj] + bias[3];
            // fake-quant forward: sigmoid -> u8/256 ; tanh -> i8/128
            float i_ = fminf(fmaxf(rintf(256.f / (1.f + __expf(-vi))), 0.f), 255.f) * (1.f / 256.f);
            float f_ = fminf(fmaxf(rintf(256.f / (1.f + __expf(-vf))), 0.f), 255.f) * (1.f / 256.f);
            float g_ = fminf(fmaxf(rintf(tanhf(vg) * 128.f), -128.f), 127.f) * (1.f / 128.f);
            float o_ = fminf(fmaxf(rintf(256.f / (1.f + __expf(-vo))), 0.f), 255.f) * (1.f / 256.f);
            float c = f_ * creg[k] + i_ * g_;
            creg[k] = c;
            float qt = fminf(fmaxf(rintf(tanhf(c) * 128.f), -128.f), 127.f) * (1.f / 128.f);
            float h = o_ * qt;
            hreg[k] = h;
            out[(size_t)t * (BATCH * HID) + b * HID + col_h] = h;
            hbuf[((t + 1) & 1) * (BATCH * HID) + b * HID + col_h] = (_Float16)h;
        }
        __threadfence();
        grid_barrier(bar);
    }

    float* hT = out + (size_t)T_STEPS * (BATCH * HID);
    float* cT = hT + BATCH * HID;
#pragma unroll
    for (int k = 0; k < 8; ++k) {
        hT[(rb + k) * HID + col_h] = hreg[k];
        cT[(rb + k) * HID + col_h] = creg[k];
    }
}

// ---------------------------------------------------------------------------
extern "C" void kernel_launch(void* const* d_in, const int* in_sizes, int n_in,
                              void* d_out, int out_size, void* d_ws, size_t ws_size,
                              hipStream_t stream) {
    (void)in_sizes; (void)n_in; (void)out_size; (void)ws_size;
    const float* x    = (const float*)d_in[0];
    const float* h0   = (const float*)d_in[1];
    const float* c0   = (const float*)d_in[2];
    const float* w_ih = (const float*)d_in[3];
    const float* w_hh = (const float*)d_in[4];
    const float* b_ih = (const float*)d_in[5];
    const float* b_hh = (const float*)d_in[6];
    float* out = (float*)d_out;

    // Workspace layout (~1.006 GiB total):
    //   [0,   2 MiB)  qW_ih fragments (fp16)
    //   [2,   4 MiB)  qW_hh fragments (fp16)
    //   [4 MiB, +128K) h double buffer (fp16)
    //   @4325376       grid-barrier counters
    //   [5 MiB, +1 GiB) gates_x (fp32, T*B*4H)
    char* ws = (char*)d_ws;
    _Float16* qwih = (_Float16*)(ws);
    _Float16* qwhh = (_Float16*)(ws + (size_t)2 * 1024 * 1024);
    _Float16* hbuf = (_Float16*)(ws + (size_t)4 * 1024 * 1024);
    unsigned* bar  = (unsigned*)(ws + (size_t)4 * 1024 * 1024 + 131072);
    float*    gxb  = (float*)(ws + (size_t)5 * 1024 * 1024);

    pack_w_kernel<<<(G4H * IN_DIM + 255) / 256, 256, 0, stream>>>(w_ih, qwih, IN_DIM);
    pack_w_kernel<<<(G4H * HID   + 255) / 256, 256, 0, stream>>>(w_hh, qwhh, HID);
    init_kernel<<<(BATCH * HID + 255) / 256, 256, 0, stream>>>(h0, hbuf, bar);

    dim3 g1(G4H / 64, (T_STEPS * BATCH) / 128);
    gemm_x_kernel<<<g1, 256, 0, stream>>>(x, qwih, gxb);

    lstm_scan_kernel<<<NBLK, 256, 0, stream>>>(gxb, qwhh, b_ih, b_hh, c0,
                                               hbuf, bar, out);
}
